// TinyViTBlock_27960237097511
// MI455X (gfx1250) — matrix operand order repaired and checked by reading, not compiled, not run on hardware
//
#include <hip/hip_runtime.h>
#include <hip/hip_bf16.h>

typedef __attribute__((ext_vector_type(16))) _Float16 v16h;
typedef __attribute__((ext_vector_type(8)))  float    v8f;

#define B_    64
#define H_    32
#define W_    32
#define C_    384
#define NH_   12
#define HD_   32
#define WS_   7
#define NWD_  5                 // ceil(32/7): windows per spatial dim (pad to 35)
#define NWIN_ (B_ * NWD_ * NWD_)   // 1600 windows
#define NTOK_ (B_ * H_ * W_)       // 65536 tokens
#define WROWS_ (NWIN_ * 49)        // 78400 window-major token rows
#define MLP_  1536
#define SCALE_ 0.17677669529663687f   // 32^-0.5

// Low 32 bits of a generic pointer into __shared__ = wave-relative LDS offset
// (ISA 10.2: LDS aperture keeps the offset in addr[31:0]).
static __device__ __forceinline__ uint32_t lds_off(const void* p) {
  return (uint32_t)(uintptr_t)p;
}

// ---------------------------------------------------------------------------
// WMMA fragment helpers (wave32; A/B in contiguous k order per lane half, C/D in ISA layout)
// ---------------------------------------------------------------------------
// A (16x32 f16, row-major in LDS, ld = row stride in halves):
//   lane<16: row=lane,    halves 0..15 = k0..15
//   lane>=16: row=lane-16, halves 0..15 = k16..31
static __device__ __forceinline__ v16h load_frag_a(const _Float16* p0, int ld) {
  int lane = threadIdx.x & 31;
  const _Float16* p = p0 + (lane & 15) * ld + ((lane < 16) ? 0 : 16);
  v16h f;
#pragma unroll
  for (int i = 0; i < 8; ++i) f[i] = p[i];
#pragma unroll
  for (int i = 0; i < 8; ++i) f[8 + i] = p[8 + i];
  return f;
}

// B (32x16 f16), staged transposed in LDS as [n][k] (ld = per-column stride):
//   lane<16: col n=lane,    halves = k0..15
//   lane>=16: col n=lane-16, halves = k16..31
static __device__ __forceinline__ v16h load_frag_b(const _Float16* p0, int ld) {
  int lane = threadIdx.x & 31;
  const _Float16* p = p0 + (lane & 15) * ld + ((lane < 16) ? 0 : 16);
  v16h f;
#pragma unroll
  for (int i = 0; i < 16; ++i) f[i] = p[i];
  return f;
}
// C/D (16x16 f32): VGPR j holds row m = j + (lane<16 ? 0 : 8), col n = lane&15.

// ---------------------------------------------------------------------------
// f32 -> f16 conversion (weights)
// ---------------------------------------------------------------------------
__global__ void f2h_kernel(const float* __restrict__ in, _Float16* __restrict__ out, int n) {
  int i = blockIdx.x * 256 + threadIdx.x;
  if (i < n) out[i] = (_Float16)in[i];
}

// ---------------------------------------------------------------------------
// LayerNorm (one 128-thread block per token row).  WINDOW=1: rows are
// window-major (pads -> LN(0) = beta, matching the reference's zero padding).
// ---------------------------------------------------------------------------
template <int WINDOW>
__global__ __launch_bounds__(128) void ln_kernel(const float* __restrict__ xin,
                                                 const float* __restrict__ g,
                                                 const float* __restrict__ bta,
                                                 _Float16* __restrict__ out) {
  int r = blockIdx.x, tid = threadIdx.x;
  size_t srow = 0;
  bool valid = true;
  if constexpr (WINDOW) {
    int win = r / 49, n = r % 49;
    int bimg = win / (NWD_ * NWD_), remw = win % (NWD_ * NWD_);
    int hh = (remw / NWD_) * WS_ + n / WS_;
    int ww = (remw % NWD_) * WS_ + n % WS_;
    valid = (hh < H_) && (ww < W_);
    srow = (size_t)bimg * (H_ * W_) + (size_t)hh * W_ + ww;
  } else {
    srow = (size_t)r;
  }
  float v0 = 0.f, v1 = 0.f, v2 = 0.f;
  if (valid) {
    const float* p = xin + srow * C_;
    v0 = p[tid]; v1 = p[tid + 128]; v2 = p[tid + 256];
  }
  __shared__ float rs[128], rq[128];
  rs[tid] = v0 + v1 + v2;
  rq[tid] = v0 * v0 + v1 * v1 + v2 * v2;
  __syncthreads();
  for (int off = 64; off; off >>= 1) {
    if (tid < off) { rs[tid] += rs[tid + off]; rq[tid] += rq[tid + off]; }
    __syncthreads();
  }
  float mean = rs[0] * (1.f / C_);
  float var  = rq[0] * (1.f / C_) - mean * mean;
  float inv  = rsqrtf(var + 1e-5f);
  _Float16* po = out + (size_t)r * C_;
  po[tid]       = (_Float16)((v0 - mean) * inv * g[tid]       + bta[tid]);
  po[tid + 128] = (_Float16)((v1 - mean) * inv * g[tid + 128] + bta[tid + 128]);
  po[tid + 256] = (_Float16)((v2 - mean) * inv * g[tid + 256] + bta[tid + 256]);
}

// ---------------------------------------------------------------------------
// 64x64-tile WMMA GEMM, 128 threads = 4 waves; wave w owns 16-col strip.
// A tile is copied global->LDS with the CDNA5 async engine (ASYNCcnt);
// B tile goes through VGPRs for the [k][n] -> [n][k] transpose; next B tile
// is prefetched into L2/WGP$ with global_prefetch.
// EPI 0: +bias -> f16             (QKV)
// EPI 1: +bias, window-reverse scatter, +residual -> f32   (proj)
// EPI 2: +bias, exact GELU -> f16 (FC1)
// EPI 3: +bias, +residual -> f32  (FC2 -> d_out)
// ---------------------------------------------------------------------------
template <int EPI>
__global__ __launch_bounds__(128) void gemm64(const _Float16* __restrict__ A,
                                              const _Float16* __restrict__ Bw,
                                              const float* __restrict__ bias,
                                              void* __restrict__ outp,
                                              const float* __restrict__ res,
                                              int M, int N, int K) {
  __shared__ __align__(16) _Float16 As[64 * 40];   // [m][k], padded
  __shared__ __align__(16) _Float16 Bs[64 * 40];   // [n][k] (transposed), padded
  int tid = threadIdx.x, lane = tid & 31, wave = tid >> 5;
  int m0 = blockIdx.y * 64, n0 = blockIdx.x * 64;
  v8f acc[4] = {};
  int arow = tid >> 1,  akoff = (tid & 1) * 16;    // 64 rows x 32 halves
  int bkrow = tid >> 2, bnoff = (tid & 3) * 16;    // 32 k-rows x 64 cols

  uint32_t ldsA = lds_off(&As[arow * 40 + akoff]);
  uint32_t gA0  = (uint32_t)(((size_t)(m0 + arow) * K + akoff) * sizeof(_Float16));

  for (int k0 = 0; k0 < K; k0 += 32) {
    // prefetch next B k-slab toward the WGP while this step computes
    if (k0 + 32 < K)
      __builtin_prefetch(Bw + (size_t)(k0 + 32 + bkrow) * N + n0 + bnoff, 0, 1);
    const uint4* bg = (const uint4*)(Bw + (size_t)(k0 + bkrow) * N + n0 + bnoff);
    uint4 b0 = bg[0], b1 = bg[1];
    __syncthreads();   // previous iteration's LDS reads complete

    // async copy of this thread's 32 bytes of the A tile straight into LDS
    uint32_t gA = gA0 + (uint32_t)(k0 * sizeof(_Float16));
    asm volatile(
        "global_load_async_to_lds_b128 %0, %1, %2\n\t"
        "global_load_async_to_lds_b128 %0, %1, %2 offset:16"
        :
        : "v"(ldsA), "v"(gA), "s"(A)
        : "memory");

    union { uint4 u[2]; _Float16 h[16]; } tb;
    tb.u[0] = b0; tb.u[1] = b1;
#pragma unroll
    for (int i = 0; i < 16; ++i) Bs[(bnoff + i) * 40 + bkrow] = tb.h[i];

    asm volatile("s_wait_asynccnt 0x0" ::: "memory");
    __syncthreads();

    v16h bf = load_frag_b(&Bs[(wave * 16) * 40], 40);
#pragma unroll
    for (int mt = 0; mt < 4; ++mt) {
      v16h af = load_frag_a(&As[(mt * 16) * 40], 40);
      acc[mt] = __builtin_amdgcn_wmma_f32_16x16x32_f16(false, af, false, bf,
                                                       (short)0, acc[mt], false, false);
    }
  }

  int ncol  = n0 + wave * 16 + (lane & 15);
  int mbase = (lane < 16) ? 0 : 8;
  float bv = bias[ncol];
#pragma unroll
  for (int mt = 0; mt < 4; ++mt) {
#pragma unroll
    for (int j = 0; j < 8; ++j) {
      int m = m0 + mt * 16 + mbase + j;
      float v = acc[mt][j] + bv;
      if constexpr (EPI == 0) {
        ((_Float16*)outp)[(size_t)m * N + ncol] = (_Float16)v;
      } else if constexpr (EPI == 2) {
        float gl = 0.5f * v * (1.0f + erff(v * 0.70710678118654752f));
        ((_Float16*)outp)[(size_t)m * N + ncol] = (_Float16)gl;
      } else if constexpr (EPI == 3) {
        size_t o = (size_t)m * N + ncol;
        ((float*)outp)[o] = v + res[o];
      } else {  // EPI == 1: window-reverse + residual
        int win = m / 49, n = m % 49;
        int bimg = win / (NWD_ * NWD_), remw = win % (NWD_ * NWD_);
        int hh = (remw / NWD_) * WS_ + n / WS_;
        int ww = (remw % NWD_) * WS_ + n % WS_;
        if (hh < H_ && ww < W_) {
          size_t o = ((size_t)bimg * (H_ * W_) + (size_t)hh * W_ + ww) * C_ + ncol;
          ((float*)outp)[o] = v + res[o];
        }
      }
    }
  }
}

// ---------------------------------------------------------------------------
// Windowed attention: one wave per (window, head).  49 tokens padded to 64.
// qkv row layout (native qkv_w columns): head h -> [h*96 .. h*96+95] = q|k|v,
// so one head's q and k rows are 128 contiguous bytes -> async-copied to LDS.
// ---------------------------------------------------------------------------
__global__ __launch_bounds__(32) void attn_kernel(const _Float16* __restrict__ qkv,
                                                  const float* __restrict__ biases,
                                                  const int* __restrict__ bidx,
                                                  _Float16* __restrict__ aw,
                                                  int n_off) {
  __shared__ __align__(16) _Float16 qs[64 * 40];   // [n][d]
  __shared__ __align__(16) _Float16 ks[64 * 40];   // [n][d]  (== k^T for B frags)
  __shared__ __align__(16) _Float16 vt[32 * 72];   // [d][key] (v transposed)
  __shared__ __align__(16) float    sc[64 * 64];   // scores f32
  __shared__ __align__(16) _Float16 ps[64 * 72];   // softmax probs f16
  int hd = blockIdx.x, win = blockIdx.y;
  int lane = threadIdx.x;

  // zero-fill pad rows 49..63 of q/k (disjoint from async targets below)
  for (int n = lane; n < 64; n += 32) {
    if (n >= 49) {
      uint4 z = {0u, 0u, 0u, 0u};
      *(uint4*)&qs[n * 40]      = z; *(uint4*)&qs[n * 40 + 8]  = z;
      *(uint4*)&qs[n * 40 + 16] = z; *(uint4*)&qs[n * 40 + 24] = z;
      *(uint4*)&ks[n * 40]      = z; *(uint4*)&ks[n * 40 + 8]  = z;
      *(uint4*)&ks[n * 40 + 16] = z; *(uint4*)&ks[n * 40 + 24] = z;
    }
  }

  // async copy valid q/k rows (64B each) straight into LDS
  for (int n = lane; n < 49; n += 32) {
    uint32_t lq  = lds_off(&qs[n * 40]);
    uint32_t lk  = lds_off(&ks[n * 40]);
    uint32_t gq  = (uint32_t)(((size_t)(win * 49 + n) * (3 * C_) + hd * 96) * sizeof(_Float16));
    uint32_t gk  = gq + 64;
    asm volatile(
        "global_load_async_to_lds_b128 %0, %2, %4\n\t"
        "global_load_async_to_lds_b128 %0, %2, %4 offset:16\n\t"
        "global_load_async_to_lds_b128 %0, %2, %4 offset:32\n\t"
        "global_load_async_to_lds_b128 %0, %2, %4 offset:48\n\t"
        "global_load_async_to_lds_b128 %1, %3, %4\n\t"
        "global_load_async_to_lds_b128 %1, %3, %4 offset:16\n\t"
        "global_load_async_to_lds_b128 %1, %3, %4 offset:32\n\t"
        "global_load_async_to_lds_b128 %1, %3, %4 offset:48"
        :
        : "v"(lq), "v"(lk), "v"(gq), "v"(gk), "s"(qkv)
        : "memory");
  }

  // v needs a transpose -> VGPR path (pads zeroed)
  for (int n = lane; n < 64; n += 32) {
    bool valid = n < 49;
    const _Float16* src =
        qkv + (size_t)(win * 49 + (valid ? n : 0)) * (3 * C_) + hd * 96 + 64;
#pragma unroll
    for (int d = 0; d < 32; ++d) {
      vt[d * 72 + n] = valid ? src[d] : (_Float16)0.f;
    }
  }

  asm volatile("s_wait_asynccnt 0x0" ::: "memory");
  __syncthreads();

  int mbase = (lane < 16) ? 0 : 8;
  int nloc  = lane & 15;

  // S = q @ k^T * scale + bias (masked)
#pragma unroll
  for (int mt = 0; mt < 4; ++mt) {
    v16h a = load_frag_a(&qs[(mt * 16) * 40], 40);
#pragma unroll
    for (int nt = 0; nt < 4; ++nt) {
      v16h b = load_frag_b(&ks[(nt * 16) * 40], 40);
      v8f c = {};
      c = __builtin_amdgcn_wmma_f32_16x16x32_f16(false, a, false, b, (short)0, c, false, false);
#pragma unroll
      for (int j = 0; j < 8; ++j) {
        int m   = mt * 16 + mbase + j;
        int col = nt * 16 + nloc;
        float val;
        if (col < 49) {
          float bb = (m < 49) ? biases[hd * n_off + bidx[m * 49 + col]] : 0.f;
          val = c[j] * SCALE_ + bb;
        } else {
          val = -1e30f;   // masked pad keys
        }
        sc[m * 64 + col] = val;
      }
    }
  }
  __syncthreads();

  // softmax rows (lane owns rows lane, lane+32)
  for (int r = lane; r < 64; r += 32) {
    if (r < 49) {
      float mx = -1e30f;
      for (int m = 0; m < 64; ++m) mx = fmaxf(mx, sc[r * 64 + m]);
      float sum = 0.f;
      for (int m = 0; m < 64; ++m) {
        float e = __expf(sc[r * 64 + m] - mx);
        sc[r * 64 + m] = e;
        sum += e;
      }
      float inv = 1.f / sum;
      for (int m = 0; m < 64; ++m) ps[r * 72 + m] = (_Float16)(sc[r * 64 + m] * inv);
    } else {
      for (int m = 0; m < 64; ++m) ps[r * 72 + m] = (_Float16)0.f;
    }
  }
  __syncthreads();

  // O = P @ V   (K = 64 keys -> 2 WMMA k-steps)
#pragma unroll
  for (int mt = 0; mt < 4; ++mt) {
#pragma unroll
    for (int nt = 0; nt < 2; ++nt) {
      v8f c = {};
#pragma unroll
      for (int kk = 0; kk < 2; ++kk) {
        v16h a = load_frag_a(&ps[(mt * 16) * 72 + kk * 32], 72);
        v16h b = load_frag_b(&vt[(nt * 16) * 72 + kk * 32], 72);
        c = __builtin_amdgcn_wmma_f32_16x16x32_f16(false, a, false, b, (short)0, c, false, false);
      }
#pragma unroll
      for (int j = 0; j < 8; ++j) {
        int m = mt * 16 + mbase + j;
        if (m < 49) {
          aw[(size_t)(win * 49 + m) * C_ + hd * HD_ + nt * 16 + nloc] = (_Float16)c[j];
        }
      }
    }
  }
}

// ---------------------------------------------------------------------------
// Depthwise 3x3 conv (pad 1) + BatchNorm.  x2 token-major [b][h*W+w][c].
// ---------------------------------------------------------------------------
__global__ __launch_bounds__(256) void conv_bn_kernel(const float* __restrict__ x2,
                                                      const float* __restrict__ cw,
                                                      const float* __restrict__ bng,
                                                      const float* __restrict__ bnb,
                                                      const float* __restrict__ bnm,
                                                      const float* __restrict__ bnv,
                                                      float* __restrict__ x3) {
  int i = blockIdx.x * 256 + threadIdx.x;
  if (i >= NTOK_ * C_) return;
  int c = i % C_;
  int t = i / C_;
  int w = t % W_;
  int h = (t / W_) % H_;
  int b = t / (H_ * W_);
  float s = 0.f;
#pragma unroll
  for (int kh = 0; kh < 3; ++kh) {
#pragma unroll
    for (int kw = 0; kw < 3; ++kw) {
      int hh = h + kh - 1, ww = w + kw - 1;
      if (hh >= 0 && hh < H_ && ww >= 0 && ww < W_) {
        s += x2[((size_t)b * (H_ * W_) + (size_t)hh * W_ + ww) * C_ + c] * cw[c * 9 + kh * 3 + kw];
      }
    }
  }
  float scl = bng[c] * rsqrtf(bnv[c] + 1e-5f);
  x3[i] = (s - bnm[c]) * scl + bnb[c];
}

// ---------------------------------------------------------------------------
// Launch
// ---------------------------------------------------------------------------
extern "C" void kernel_launch(void* const* d_in, const int* in_sizes, int n_in,
                              void* d_out, int out_size, void* d_ws, size_t ws_size,
                              hipStream_t stream) {
  (void)n_in; (void)out_size; (void)ws_size;
  const float* x       = (const float*)d_in[0];
  const float* ln1_g   = (const float*)d_in[1];
  const float* ln1_b   = (const float*)d_in[2];
  const float* qkv_w   = (const float*)d_in[3];
  const float* qkv_b   = (const float*)d_in[4];
  const float* proj_w  = (const float*)d_in[5];
  const float* proj_b  = (const float*)d_in[6];
  const float* attn_bs = (const float*)d_in[7];
  const float* conv_w  = (const float*)d_in[8];
  const float* bn_g    = (const float*)d_in[9];
  const float* bn_b    = (const float*)d_in[10];
  const float* bn_m    = (const float*)d_in[11];
  const float* bn_v    = (const float*)d_in[12];
  const float* ln2_g   = (const float*)d_in[13];
  const float* ln2_b   = (const float*)d_in[14];
  const float* fc1_w   = (const float*)d_in[15];
  const float* fc1_b   = (const float*)d_in[16];
  const float* fc2_w   = (const float*)d_in[17];
  const float* fc2_b   = (const float*)d_in[18];
  const int*   bidx    = (const int*)d_in[19];
  int n_off = in_sizes[7] / NH_;

  char* p = (char*)d_ws;
  auto alloc = [&](size_t bytes) -> void* {
    void* r = (void*)p;
    p += (bytes + 255) & ~(size_t)255;
    return r;
  };
  _Float16* xn_h    = (_Float16*)alloc((size_t)WROWS_ * C_ * 2);
  _Float16* qkv_h   = (_Float16*)alloc((size_t)WROWS_ * 3 * C_ * 2);
  _Float16* aw_h    = (_Float16*)alloc((size_t)WROWS_ * C_ * 2);
  float*    x2      = (float*)alloc((size_t)NTOK_ * C_ * 4);
  float*    x3      = (float*)alloc((size_t)NTOK_ * C_ * 4);
  _Float16* xn2_h   = (_Float16*)alloc((size_t)NTOK_ * C_ * 2);
  _Float16* h_h     = (_Float16*)alloc((size_t)NTOK_ * MLP_ * 2);
  _Float16* qkvw_h  = (_Float16*)alloc((size_t)C_ * 3 * C_ * 2);
  _Float16* projw_h = (_Float16*)alloc((size_t)C_ * C_ * 2);
  _Float16* fc1w_h  = (_Float16*)alloc((size_t)C_ * MLP_ * 2);
  _Float16* fc2w_h  = (_Float16*)alloc((size_t)MLP_ * C_ * 2);

  // weights -> f16
  f2h_kernel<<<(C_ * 3 * C_ + 255) / 256, 256, 0, stream>>>(qkv_w, qkvw_h, C_ * 3 * C_);
  f2h_kernel<<<(C_ * C_ + 255) / 256, 256, 0, stream>>>(proj_w, projw_h, C_ * C_);
  f2h_kernel<<<(C_ * MLP_ + 255) / 256, 256, 0, stream>>>(fc1_w, fc1w_h, C_ * MLP_);
  f2h_kernel<<<(MLP_ * C_ + 255) / 256, 256, 0, stream>>>(fc2_w, fc2w_h, MLP_ * C_);

  // LN1 in window-major order (pads -> beta)
  ln_kernel<1><<<WROWS_, 128, 0, stream>>>(x, ln1_g, ln1_b, xn_h);

  // QKV GEMM [78400,384] x [384,1152]
  gemm64<0><<<dim3(3 * C_ / 64, WROWS_ / 64), 128, 0, stream>>>(
      xn_h, qkvw_h, qkv_b, (void*)qkv_h, nullptr, WROWS_, 3 * C_, C_);

  // windowed attention
  attn_kernel<<<dim3(NH_, NWIN_), 32, 0, stream>>>(qkv_h, attn_bs, bidx, aw_h, n_off);

  // proj GEMM + window-reverse + residual -> x2
  gemm64<1><<<dim3(C_ / 64, WROWS_ / 64), 128, 0, stream>>>(
      aw_h, projw_h, proj_b, (void*)x2, x, WROWS_, C_, C_);

  // depthwise conv + BN -> x3
  conv_bn_kernel<<<(NTOK_ * C_) / 256, 256, 0, stream>>>(x2, conv_w, bn_g, bn_b, bn_m, bn_v, x3);

  // LN2
  ln_kernel<0><<<NTOK_, 128, 0, stream>>>(x3, ln2_g, ln2_b, xn2_h);

  // FC1 + GELU
  gemm64<2><<<dim3(MLP_ / 64, NTOK_ / 64), 128, 0, stream>>>(
      xn2_h, fc1w_h, fc1_b, (void*)h_h, nullptr, NTOK_, MLP_, C_);

  // FC2 + residual -> d_out (f32)
  gemm64<3><<<dim3(C_ / 64, NTOK_ / 64), 128, 0, stream>>>(
      h_h, fc2w_h, fc2_b, d_out, x3, NTOK_, C_, MLP_);
}
